// MSAAttention_23596550324276
// MI455X (gfx1250) — compile-verified, matrix operand order checked by reading, not run on hardware
//
#include <hip/hip_runtime.h>
#include <hip/hip_bf16.h>

typedef __attribute__((ext_vector_type(16))) __bf16 v16bf;
typedef __attribute__((ext_vector_type(8)))  float  v8f;

#define NUM_SEQ 512
#define NUM_RES 768
#define NHEAD 8
#define NDIM  4096   // NUM_SEQ * 8 (val_dim)
#define KT_RES 24    // 768/32 k-steps
#define QT16   48    // 768/16
#define NT16   256   // 4096/16
#define RT_TOT 24576 // 393216/16 row tiles

// CDNA5 wave32 wmma 16x16x32 bf16 fragment mapping:
// element (m,k) of a 16x32 tile lives at lane = m + 16*((k>>3)&1),
// vector element e = ((k>>4)<<3) | (k&7).  One tile = 32 lanes x 16 bf16 = 1KB.
__device__ inline int frag_idx(int m, int k) {
  return (m + (((k >> 3) & 1) << 4)) * 16 + (((k >> 4) << 3) | (k & 7));
}
__device__ inline v16bf frag_ld(const __bf16* __restrict__ t) {
  return ((const v16bf*)t)[threadIdx.x & 31];
}
__device__ inline v8f wmma_bf16(v16bf a, v16bf b, v8f c) {
  return __builtin_amdgcn_wmma_f32_16x16x32_bf16(false, a, false, b, (short)0, c, false, false);
}

// ============ Kernel A: LN(msa) -> V (B-frag order) and gate (A-frag order) ============
__global__ __launch_bounds__(256) void k_msa_ln_v_gate(
    const float* __restrict__ msa, const float* __restrict__ g,
    const float* __restrict__ be, const float* __restrict__ w_v,
    const float* __restrict__ w_gate,
    __bf16* __restrict__ Vf, __bf16* __restrict__ Gf)
{
  __shared__ alignas(32) __bf16 sWv[8 * 512];   // [kt2][nt4] fragment tiles
  __shared__ alignas(32) __bf16 sWg[8 * 512];
  __shared__ alignas(32) __bf16 sX[8][2 * 512]; // per-wave [ct2] fragment tiles
  for (int i = threadIdx.x; i < 4096; i += 256) {
    int k = i >> 6, n = i & 63;
    int d = (((k >> 5) << 2) | (n >> 4)) * 512 + frag_idx(n & 15, k & 31);
    sWv[d] = (__bf16)w_v[n * 64 + k];
    sWg[d] = (__bf16)w_gate[n * 64 + k];
  }
  __syncthreads();
  int wave = threadIdx.x >> 5, lane = threadIdx.x & 31;
  long r0 = ((long)blockIdx.x * 8 + wave) * 16;   // row = b*768 + k_res
  __bf16* X = sX[wave];
  int c0 = lane * 2;
  float g0 = g[c0], g1 = g[c0 + 1], b0 = be[c0], b1 = be[c0 + 1];
  for (int m = 0; m < 16; ++m) {
    const float2 xv = ((const float2*)(msa + (r0 + m) * 64))[lane];
    float s = xv.x + xv.y;
#pragma unroll
    for (int o = 16; o; o >>= 1) s += __shfl_xor(s, o);
    float mu = s * (1.f / 64.f);
    float d0 = xv.x - mu, d1 = xv.y - mu;
    float v2 = d0 * d0 + d1 * d1;
#pragma unroll
    for (int o = 16; o; o >>= 1) v2 += __shfl_xor(v2, o);
    float rs = rsqrtf(v2 * (1.f / 64.f) + 1e-5f);
    int d = (c0 >> 5) * 512 + frag_idx(m, c0 & 31);  // c0 even -> c0+1 is d+1
    X[d]     = (__bf16)(d0 * rs * g0 + b0);
    X[d + 1] = (__bf16)(d1 * rs * g1 + b1);
  }
  __syncthreads();
  v16bf a0 = frag_ld(X), a1 = frag_ld(X + 512);
  long bidx = r0 / NUM_RES;
  int  k0r  = (int)(r0 % NUM_RES);
  long rt   = r0 >> 4;
  int  hi   = (lane >> 4) << 3;
#pragma unroll
  for (int t = 0; t < 4; ++t) {
    v8f av = {}, ag = {};
    av = wmma_bf16(a0, frag_ld(sWv + t * 512), av);
    av = wmma_bf16(a1, frag_ld(sWv + (4 + t) * 512), av);
    ag = wmma_bf16(a0, frag_ld(sWg + t * 512), ag);
    ag = wmma_bf16(a1, frag_ld(sWg + (4 + t) * 512), ag);
    int o = t * 16 + (lane & 15);
    int h = o >> 3, c = o & 7;
    int ct = o >> 5, cc = o & 31;
#pragma unroll
    for (int r = 0; r < 8; ++r) {
      int m = r + hi;
      int k_res = k0r + m;
      int n = (int)(bidx * 8) + c;
      // Vf: B-fragment tiles [h][nt16][kt]
      Vf[(((long)h * NT16 + (n >> 4)) * KT_RES + (k_res >> 5)) * 512 +
         frag_idx(n & 15, k_res & 31)] = (__bf16)av[r];
      float x = ag[r];
      // Gf: A-fragment tiles [rt][ct2]
      Gf[(rt * 2 + ct) * 512 + frag_idx(m, cc)] =
          (__bf16)(1.f / (1.f + expf(-x)));
    }
  }
}

// ============ Kernel B: LN(pair) -> logits[h][q][k] (natural f32) ============
__global__ __launch_bounds__(256) void k_pair_logits(
    const float* __restrict__ pair, const float* __restrict__ g,
    const float* __restrict__ be, const float* __restrict__ w_logits,
    float* __restrict__ logits)
{
  __shared__ alignas(32) __bf16 sW[4 * 512];     // [kt4] frag tiles, heads padded
  __shared__ alignas(32) __bf16 sX[8][4 * 512];  // per-wave [ct4]
  for (int i = threadIdx.x; i < 4 * 512; i += 256) sW[i] = (__bf16)0.f;
  __syncthreads();
  for (int i = threadIdx.x; i < 128 * 8; i += 256) {
    int k = i >> 3, n = i & 7;
    sW[(k >> 5) * 512 + frag_idx(n, k & 31)] = (__bf16)w_logits[n * 128 + k];
  }
  __syncthreads();
  int wave = threadIdx.x >> 5, lane = threadIdx.x & 31;
  long r0 = ((long)blockIdx.x * 8 + wave) * 16;   // pixel = q*768 + k
  __bf16* X = sX[wave];
  int c4 = lane * 4;
  float gg[4] = { g[c4], g[c4 + 1], g[c4 + 2], g[c4 + 3] };
  float bb[4] = { be[c4], be[c4 + 1], be[c4 + 2], be[c4 + 3] };
  for (int m = 0; m < 16; ++m) {
    const float4 xv = ((const float4*)(pair + (r0 + m) * 128))[lane];
    float s = xv.x + xv.y + xv.z + xv.w;
#pragma unroll
    for (int o = 16; o; o >>= 1) s += __shfl_xor(s, o);
    float mu = s * (1.f / 128.f);
    float d[4] = { xv.x - mu, xv.y - mu, xv.z - mu, xv.w - mu };
    float v2 = d[0] * d[0] + d[1] * d[1] + d[2] * d[2] + d[3] * d[3];
#pragma unroll
    for (int o = 16; o; o >>= 1) v2 += __shfl_xor(v2, o);
    float rs = rsqrtf(v2 * (1.f / 128.f) + 1e-5f);
    int dd = (c4 >> 5) * 512 + frag_idx(m, c4 & 31);  // c4%8 in {0,4}: 4 consecutive
#pragma unroll
    for (int j = 0; j < 4; ++j)
      X[dd + j] = (__bf16)(d[j] * rs * gg[j] + bb[j]);
  }
  __syncthreads();
  v8f acc = {};
#pragma unroll
  for (int kt = 0; kt < 4; ++kt)
    acc = wmma_bf16(frag_ld(X + kt * 512), frag_ld(sW + kt * 512), acc);
  int n = lane & 15, hi = (lane >> 4) << 3;
  if (n < NHEAD) {
#pragma unroll
    for (int r = 0; r < 8; ++r)
      logits[(long)n * (NUM_RES * NUM_RES) + r0 + r + hi] = acc[r];
  }
}

// ============ Kernel C0: mask bias ============
__global__ void k_bias(const float* __restrict__ mask, float* __restrict__ bias) {
  int k = blockIdx.x * blockDim.x + threadIdx.x;
  if (k >= NUM_RES) return;
  float m = -3.0e38f;
  for (int b = 0; b < NUM_SEQ; ++b) m = fmaxf(m, mask[b * NUM_RES + k]);
  bias[k] = 1e9f * (m - 1.f);
}

// ============ Kernel C: softmax over k -> Wf in A-fragment order ============
__global__ __launch_bounds__(256) void k_softmax(
    const float* __restrict__ logits, const float* __restrict__ bias,
    __bf16* __restrict__ Wf)
{
  long row = blockIdx.x;                    // h*768 + q
  int h = (int)(row / NUM_RES), q = (int)(row % NUM_RES);
  const float* in = logits + row * NUM_RES;
  __shared__ float red[8];
  int t = threadIdx.x, lane = t & 31, wave = t >> 5;
  float x0 = in[t] + bias[t];
  float x1 = in[t + 256] + bias[t + 256];
  float x2 = in[t + 512] + bias[t + 512];
  float mx = fmaxf(fmaxf(x0, x1), x2);
#pragma unroll
  for (int o = 16; o; o >>= 1) mx = fmaxf(mx, __shfl_xor(mx, o));
  if (lane == 0) red[wave] = mx;
  __syncthreads();
  mx = red[0];
#pragma unroll
  for (int w = 1; w < 8; ++w) mx = fmaxf(mx, red[w]);
  float e0 = expf(x0 - mx), e1 = expf(x1 - mx), e2 = expf(x2 - mx);
  float s = e0 + e1 + e2;
#pragma unroll
  for (int o = 16; o; o >>= 1) s += __shfl_xor(s, o);
  __syncthreads();
  if (lane == 0) red[wave] = s;
  __syncthreads();
  s = 0.f;
#pragma unroll
  for (int w = 0; w < 8; ++w) s += red[w];
  float inv = 1.f / s;
  __bf16* base = Wf + ((long)h * QT16 + (q >> 4)) * KT_RES * 512;
  int m = q & 15;
#pragma unroll
  for (int j = 0; j < 3; ++j) {
    int k = t + j * 256;
    float v = (j == 0 ? e0 : (j == 1 ? e1 : e2)) * inv;
    base[(k >> 5) * 512 + frag_idx(m, k & 31)] = (__bf16)v;
  }
}

// ============ Kernel D: per-head GEMM v_avg = W[h] @ V[h], no LDS ============
// one wave owns a 32(M=q) x 64(N=b*8+c) tile: 2 A-frags x 4 B-frags, 8 acc,
// 8 wmma per k-step x 24 k-steps.  All operands fragment-ordered in L2.
__global__ __launch_bounds__(256) void k_attn_gemm(
    const __bf16* __restrict__ Wf, const __bf16* __restrict__ Vf,
    __bf16* __restrict__ Af)
{
  int wave = threadIdx.x >> 5, lane = threadIdx.x & 31;
  int w = blockIdx.x * 8 + wave;            // 12288 wave-tiles
  int h = w / 1536, rem = w % 1536;
  int qt32 = rem / 64, nt64 = rem % 64;
  const __bf16* Wb = Wf + ((long)h * QT16 + qt32 * 2) * KT_RES * 512;
  const __bf16* Vb = Vf + ((long)h * NT16 + nt64 * 4) * KT_RES * 512;
  v8f acc[2][4] = {};
  for (int kt = 0; kt < KT_RES; ++kt) {
    v16bf a0 = frag_ld(Wb + kt * 512);
    v16bf a1 = frag_ld(Wb + (KT_RES + kt) * 512);
#pragma unroll
    for (int j = 0; j < 4; ++j) {
      v16bf b = frag_ld(Vb + (j * KT_RES + kt) * 512);
      acc[0][j] = wmma_bf16(a0, b, acc[0][j]);
      acc[1][j] = wmma_bf16(a1, b, acc[1][j]);
    }
  }
  int hi = (lane >> 4) << 3;
#pragma unroll
  for (int mi = 0; mi < 2; ++mi) {
#pragma unroll
    for (int j = 0; j < 4; ++j) {
      int n = nt64 * 64 + j * 16 + (lane & 15);
      int b = n >> 3, c = n & 7;
      int o = h * 8 + c;
      int ct = o >> 5, cc = o & 31;
#pragma unroll
      for (int r = 0; r < 8; ++r) {
        int q = qt32 * 32 + mi * 16 + r + hi;
        long rt = ((long)b * NUM_RES + q) >> 4;
        Af[(rt * 2 + ct) * 512 + frag_idx(q & 15, cc)] = (__bf16)acc[mi][j][r];
      }
    }
  }
}

// ============ Kernel E: out = (v_avg * gate) @ w_out^T ============
__global__ __launch_bounds__(256) void k_gate_out(
    const __bf16* __restrict__ Af, const __bf16* __restrict__ Gf,
    const float* __restrict__ w_out, float* __restrict__ out)
{
  __shared__ alignas(32) __bf16 sW[8 * 512];   // [kt2][nt4] frag tiles
  for (int i = threadIdx.x; i < 4096; i += 256) {
    int k = i >> 6, n = i & 63;
    sW[(((k >> 5) << 2) | (n >> 4)) * 512 + frag_idx(n & 15, k & 31)] =
        (__bf16)w_out[n * 64 + k];
  }
  __syncthreads();
  int wave = threadIdx.x >> 5, lane = threadIdx.x & 31;
  long rt = (long)blockIdx.x * 8 + wave;
  v16bf x[2];
#pragma unroll
  for (int ct = 0; ct < 2; ++ct) {
    v16bf va = frag_ld(Af + (rt * 2 + ct) * 512);
    v16bf vg = frag_ld(Gf + (rt * 2 + ct) * 512);
#pragma unroll
    for (int e = 0; e < 16; ++e)
      x[ct][e] = (__bf16)((float)va[e] * (float)vg[e]);
  }
  int hi = (lane >> 4) << 3;
#pragma unroll
  for (int t = 0; t < 4; ++t) {
    v8f acc = {};
    acc = wmma_bf16(x[0], frag_ld(sW + t * 512), acc);
    acc = wmma_bf16(x[1], frag_ld(sW + (4 + t) * 512), acc);
    int o = t * 16 + (lane & 15);
#pragma unroll
    for (int r = 0; r < 8; ++r)
      out[(rt * 16 + r + hi) * 64 + o] = acc[r];
  }
}

extern "C" void kernel_launch(void* const* d_in, const int* in_sizes, int n_in,
                              void* d_out, int out_size, void* d_ws, size_t ws_size,
                              hipStream_t stream) {
  const float* msa      = (const float*)d_in[0];
  const float* msa_mask = (const float*)d_in[1];
  const float* pair     = (const float*)d_in[2];
  const float* ln_msa_g = (const float*)d_in[3];
  const float* ln_msa_b = (const float*)d_in[4];
  const float* ln_pair_g= (const float*)d_in[5];
  const float* ln_pair_b= (const float*)d_in[6];
  const float* w_logits = (const float*)d_in[7];
  const float* w_v      = (const float*)d_in[8];
  const float* w_gate   = (const float*)d_in[9];
  const float* w_out    = (const float*)d_in[10];
  float* out = (float*)d_out;

  // workspace: all fragment-ordered bf16 tile buffers (1KB tiles), ~171 MB
  char* ws = (char*)d_ws;
  size_t off = 0;
  __bf16* Vf     = (__bf16*)(ws + off); off += (size_t)NHEAD * NT16 * KT_RES * 512 * 2;   // 50.3 MB
  __bf16* Gf     = (__bf16*)(ws + off); off += (size_t)RT_TOT * 2 * 512 * 2;              // 50.3 MB
  float*  logits = (float*) (ws + off); off += (size_t)NHEAD * NUM_RES * NUM_RES * 4;     // 18.9 MB
  __bf16* Wf     = (__bf16*)(ws + off); off += (size_t)NHEAD * QT16 * KT_RES * 512 * 2;   //  9.4 MB
  __bf16* Af     = (__bf16*)(ws + off); off += (size_t)RT_TOT * 2 * 512 * 2;              // 50.3 MB
  float*  bias   = (float*) (ws + off); off += NUM_RES * 4;

  k_msa_ln_v_gate<<<(NUM_SEQ * NUM_RES) / 128, 256, 0, stream>>>(
      msa, ln_msa_g, ln_msa_b, w_v, w_gate, Vf, Gf);
  k_pair_logits<<<(NUM_RES * NUM_RES) / 128, 256, 0, stream>>>(
      pair, ln_pair_g, ln_pair_b, w_logits, logits);
  k_bias<<<3, 256, 0, stream>>>(msa_mask, bias);
  k_softmax<<<NHEAD * NUM_RES, 256, 0, stream>>>(logits, bias, Wf);
  k_attn_gemm<<<1536, 256, 0, stream>>>(Wf, Vf, Af);
  k_gate_out<<<(NUM_SEQ * NUM_RES) / 128, 256, 0, stream>>>(Af, Gf, w_out, out);
}
